// SDgraphConv_49692771615249
// MI455X (gfx1250) — matrix operand order repaired and checked by reading, not compiled, not run on hardware
//
#include <hip/hip_runtime.h>

typedef __attribute__((ext_vector_type(16))) _Float16 v16h;
typedef __attribute__((ext_vector_type(8)))  _Float16 v8h;
typedef __attribute__((ext_vector_type(8)))  float    v8f;

#define NN 10000
#define DD 64
#define EE 320000
#define KK 16
#define NEDYN (NN * 17)   // 16 kNN neighbors + self loop per node

// ---------------- WMMA helpers (CDNA5, wave32) ----------------

__device__ __forceinline__ v8f wmma16(v16h a, v16h b, v8f c) {
  // D = A(16x32 f16) x B(32x16 f16) + C(16x16 f32)
  return __builtin_amdgcn_wmma_f32_16x16x32_f16(false, a, false, b, (short)0, c,
                                                false, false);
}

// A-matrix 16x32 f16 (ISA 7.12.2): lane holds row M = lane&15; its 16 halfs
// are two contiguous K-runs of 8 (code's order, not ISA's): k = kt*32 + (lane>>4)*16 + run*8 + [0..8).
// B-matrix 32x16 f16: lane holds col N = lane&15; its 16 halfs are one
// contiguous K-run: k = kt*32 + (lane>>4)*16 + [0..16).

__device__ __forceinline__ v8f vz8() {
  v8f z = {0.f, 0.f, 0.f, 0.f, 0.f, 0.f, 0.f, 0.f};
  return z;
}
__device__ __forceinline__ v8h z8h() {
  v8h z = {(_Float16)0.f, (_Float16)0.f, (_Float16)0.f, (_Float16)0.f,
           (_Float16)0.f, (_Float16)0.f, (_Float16)0.f, (_Float16)0.f};
  return z;
}
__device__ __forceinline__ v16h cat16(v8h lo, v8h hi) {
  return __builtin_shufflevector(lo, hi, 0, 1, 2, 3, 4, 5, 6, 7, 8, 9, 10, 11,
                                 12, 13, 14, 15);
}
__device__ __forceinline__ v8h cvt8(float4 a, float4 b) {
  v8h r;
  r[0] = (_Float16)a.x; r[1] = (_Float16)a.y;
  r[2] = (_Float16)a.z; r[3] = (_Float16)a.w;
  r[4] = (_Float16)b.x; r[5] = (_Float16)b.y;
  r[6] = (_Float16)b.z; r[7] = (_Float16)b.w;
  return r;
}
// 8 consecutive floats (16B aligned) -> v8h
__device__ __forceinline__ v8h load8f(const float* __restrict__ p) {
  float4 a = *(const float4*)p;
  float4 b = *(const float4*)(p + 4);
  return cvt8(a, b);
}
// (pa[0..8) - pb[0..8)) -> v8h
__device__ __forceinline__ v8h sub8f(const float* __restrict__ pa,
                                     const float* __restrict__ pb) {
  float4 a0 = *(const float4*)pa, a1 = *(const float4*)(pa + 4);
  float4 b0 = *(const float4*)pb, b1 = *(const float4*)(pb + 4);
  float4 d0 = make_float4(a0.x - b0.x, a0.y - b0.y, a0.z - b0.z, a0.w - b0.w);
  float4 d1 = make_float4(a1.x - b1.x, a1.y - b1.y, a1.z - b1.z, a1.w - b1.w);
  return cvt8(d0, d1);
}
__device__ __forceinline__ float frelu(float x) { return x > 0.f ? x : 0.f; }

// ---------------- zero init ----------------

__global__ void zero_kernel(float* __restrict__ p, int n) {
  int i = blockIdx.x * 256 + threadIdx.x;
  if (i < n) p[i] = 0.f;
}

// ---------------- brute-force kNN (k=16) on location_info (N x 4) ----------------

__global__ void knn_kernel(const float* __restrict__ loc, int* __restrict__ knn,
                           int n) {
  __shared__ float cd[256 * 16];
  __shared__ int   ci[256 * 16];
  __shared__ float redD[256];
  __shared__ int   redI[256];
  const int i   = blockIdx.x;
  const int tid = threadIdx.x;
  const float l0 = loc[i * 4 + 0], l1 = loc[i * 4 + 1];
  const float l2 = loc[i * 4 + 2], l3 = loc[i * 4 + 3];

  float bd[16];
  int   bi[16];
#pragma unroll
  for (int t = 0; t < 16; ++t) { bd[t] = __builtin_inff(); bi[t] = -1; }

  for (int j = tid; j < n; j += 256) {
    if (j == i) continue;
    float d0 = loc[j * 4 + 0] - l0, d1 = loc[j * 4 + 1] - l1;
    float d2 = loc[j * 4 + 2] - l2, d3 = loc[j * 4 + 3] - l3;
    float dist = d0 * d0 + d1 * d1 + d2 * d2 + d3 * d3;
    if (dist < bd[15]) {
      int p = 15;
      while (p > 0 && bd[p - 1] > dist) {
        bd[p] = bd[p - 1]; bi[p] = bi[p - 1]; --p;
      }
      bd[p] = dist; bi[p] = j;
    }
  }
#pragma unroll
  for (int t = 0; t < 16; ++t) {
    cd[tid * 16 + t] = bd[t];
    ci[tid * 16 + t] = bi[t];
  }
  __syncthreads();

  for (int round = 0; round < 16; ++round) {
    float md = __builtin_inff();
    int   mi = -1;
    for (int idx = tid; idx < 256 * 16; idx += 256)
      if (cd[idx] < md) { md = cd[idx]; mi = idx; }
    redD[tid] = md; redI[tid] = mi;
    __syncthreads();
    for (int off = 128; off > 0; off >>= 1) {
      if (tid < off && redD[tid + off] < redD[tid]) {
        redD[tid] = redD[tid + off];
        redI[tid] = redI[tid + off];
      }
      __syncthreads();
    }
    if (tid == 0) {
      int widx = redI[0];
      knn[i * 16 + round] = (widx >= 0) ? ci[widx] : i;
      if (widx >= 0) cd[widx] = __builtin_inff();
    }
    __syncthreads();
  }
}

// ---------------- static conv message + segment-max ----------------
// m = relu(Wm @ [edge_attr(16); h[src]-h[dst](64)] + bm), agg[dst] = max(agg, m)
// Wm kept in LDS in native [n][k] layout (k padded 80->96) so each B-fragment
// is one aligned v16h LDS load.

__global__ void static_msg_kernel(const float* __restrict__ h,
                                  const float* __restrict__ eattr,
                                  const int* __restrict__ srcv,
                                  const int* __restrict__ dstv,
                                  const float* __restrict__ wm,  // 64 x 80
                                  const float* __restrict__ bm,  // 64
                                  float* __restrict__ agg,       // N x 64, zeroed
                                  int E) {
  __shared__ alignas(32) _Float16 bw[64 * 96];  // [n][k], k padded to 96
  const int tid = threadIdx.x;
  for (int idx = tid; idx < 64 * 96; idx += 256) {
    int nf = idx / 96, k = idx - nf * 96;
    bw[idx] = (_Float16)((k < 80) ? wm[nf * 80 + k] : 0.f);
  }
  __syncthreads();

  const int lane = tid & 31, wv = tid >> 5;
  const int tileBase = blockIdx.x * 128 + wv * 16;
  if (tileBase >= E) return;  // wave-uniform, no barriers below

  const int g    = lane >> 4;
  const int erow = tileBase + (lane & 15);
  const bool ev  = erow < E;
  const int es   = ev ? srcv[erow] : 0;
  const int ed   = ev ? dstv[erow] : 0;

  // Build A fragments: X = [edge_attr | h[src]-h[dst] | 0-pad], K = 96 = 3x32
  v16h A[3];
#pragma unroll
  for (int kt = 0; kt < 3; ++kt) {
    v8h run[2];
#pragma unroll
    for (int rr = 0; rr < 2; ++rr) {
      int k0 = kt * 32 + g * 16 + rr * 8;  // 8-run, never crosses a segment
      v8h vv = z8h();
      if (ev) {
        if (k0 < 16)
          vv = load8f(&eattr[(size_t)erow * 16 + k0]);
        else if (k0 < 80)
          vv = sub8f(&h[(size_t)es * 64 + (k0 - 16)],
                     &h[(size_t)ed * 64 + (k0 - 16)]);
      }
      run[rr] = vv;
    }
    A[kt] = cat16(run[0], run[1]);
  }

  int drow[8]; bool dval[8];
#pragma unroll
  for (int r = 0; r < 8; ++r) {
    int e   = tileBase + r + 8 * g;
    dval[r] = e < E;
    drow[r] = dval[r] ? dstv[e] : 0;
  }

  const int nf = lane & 15;
#pragma unroll
  for (int nt = 0; nt < 4; ++nt) {
    v8f acc = vz8();
#pragma unroll
    for (int kt = 0; kt < 3; ++kt) {
      v16h b = *(const v16h*)&bw[(nt * 16 + nf) * 96 + kt * 32 + g * 16];
      acc = wmma16(A[kt], b, acc);
    }
    float bias = bm[nt * 16 + nf];
#pragma unroll
    for (int r = 0; r < 8; ++r) {
      if (dval[r]) {
        float val = frelu(acc[r] + bias);
        atomicMax((unsigned int*)&agg[(size_t)drow[r] * 64 + nt * 16 + nf],
                  __float_as_uint(val));
      }
    }
  }
}

// ---------------- generic fused GEMM: out = act(cat(segs) @ W^T + b) ----------------
// Segments are 64-wide tables (N x 64). K = 64*nsegs (<=384). W kept in LDS in
// native [n][k] layout -> B-fragment = one v16h LDS load.

struct Segs { const float* p[6]; };

__global__ void gemm_relu_kernel(Segs segs,
                                 const float* __restrict__ w,   // 64 x Kdim
                                 const float* __restrict__ bias,
                                 float* __restrict__ out,       // N x 64
                                 int n, int Kdim, int do_relu) {
  __shared__ alignas(32) _Float16 wT[64 * 384];  // [n][k]
  const int tid = threadIdx.x;
  const int total = 64 * Kdim;
  for (int idx = tid * 4; idx < total; idx += 256 * 4) {
    float4 f = *(const float4*)&w[idx];
    wT[idx + 0] = (_Float16)f.x; wT[idx + 1] = (_Float16)f.y;
    wT[idx + 2] = (_Float16)f.z; wT[idx + 3] = (_Float16)f.w;
  }
  __syncthreads();

  const int lane = tid & 31, wv = tid >> 5;
  const int tileBase = blockIdx.x * 128 + wv * 16;
  if (tileBase >= n) return;  // wave-uniform, no barriers below

  const int g   = lane >> 4;
  const int row = tileBase + (lane & 15);
  const bool rv = row < n;
  const int nf  = lane & 15;

  v8f acc[4];
#pragma unroll
  for (int t = 0; t < 4; ++t) acc[t] = vz8();

  const int nK32 = Kdim >> 5;
  for (int kt = 0; kt < nK32; ++kt) {
    v8h run[2];
#pragma unroll
    for (int rr = 0; rr < 2; ++rr) {
      int k0 = kt * 32 + g * 16 + rr * 8;  // 8-run within one 64-wide segment
      run[rr] = rv ? load8f(&segs.p[k0 >> 6][(size_t)row * 64 + (k0 & 63)])
                   : z8h();
    }
    v16h a = cat16(run[0], run[1]);
#pragma unroll
    for (int nt = 0; nt < 4; ++nt) {
      v16h b = *(const v16h*)&wT[(nt * 16 + nf) * Kdim + kt * 32 + g * 16];
      acc[nt] = wmma16(a, b, acc[nt]);
    }
  }

#pragma unroll
  for (int nt = 0; nt < 4; ++nt) {
    float bb = bias[nt * 16 + nf];
#pragma unroll
    for (int r = 0; r < 8; ++r) {
      int orow = tileBase + r + 8 * g;
      if (orow < n) {
        float val = acc[nt][r] + bb;
        if (do_relu) val = frelu(val);
        out[(size_t)orow * 64 + nt * 16 + nf] = val;
      }
    }
  }
}

// ---------------- dynamic conv (kNN EdgeConv) ----------------
// For edge (i <- s): m = relu(relu([xi; xs-xi](128) @ W1^T) @ W2^T), g[i]=max.
// 128 threads = 4 waves x 16-edge tiles. W1 (32KB) + W2 (16KB) + t1 (16KB) in
// LDS = 64KB/block. Edge id e: i=e/17, j=e%17 (j==16 -> self loop).

__global__ void dyn_msg_kernel(const float* __restrict__ h,
                               const int* __restrict__ knn,
                               const float* __restrict__ w1,  // 128 x 128
                               const float* __restrict__ b1,  // 128
                               const float* __restrict__ w2,  // 64 x 128
                               const float* __restrict__ b2,  // 64
                               float* __restrict__ g,         // N x 64, zeroed
                               int n) {
  __shared__ alignas(32) _Float16 w1T[128 * 128];     // [n][k]
  __shared__ alignas(32) _Float16 w2T[64 * 128];      // [n][k]
  __shared__ alignas(32) _Float16 t1s[4 * 16 * 128];  // per-wave 16x128, [m][k]
  const int tid = threadIdx.x;
  for (int idx = tid * 4; idx < 128 * 128; idx += 128 * 4) {
    float4 f = *(const float4*)&w1[idx];
    w1T[idx + 0] = (_Float16)f.x; w1T[idx + 1] = (_Float16)f.y;
    w1T[idx + 2] = (_Float16)f.z; w1T[idx + 3] = (_Float16)f.w;
  }
  for (int idx = tid * 4; idx < 64 * 128; idx += 128 * 4) {
    float4 f = *(const float4*)&w2[idx];
    w2T[idx + 0] = (_Float16)f.x; w2T[idx + 1] = (_Float16)f.y;
    w2T[idx + 2] = (_Float16)f.z; w2T[idx + 3] = (_Float16)f.w;
  }
  __syncthreads();

  const int lane = tid & 31, wv = tid >> 5;
  const int gg   = lane >> 4;
  const int tileBase = blockIdx.x * 64 + wv * 16;
  const int NE = n * 17;
  const int m  = lane & 15;
  const int e  = tileBase + m;
  const bool ev = e < NE;
  const int ti = ev ? (e / 17) : 0;
  const int jj = ev ? (e % 17) : 16;
  const int sv = (jj < 16) ? knn[ti * 16 + jj] : ti;
  const int nf = lane & 15;

  // ---- GEMM1: t1 = relu([xi; xs-xi] @ W1^T + b1), 16x128 ----
  v8f acc[8];
#pragma unroll
  for (int t = 0; t < 8; ++t) acc[t] = vz8();

#pragma unroll
  for (int kt = 0; kt < 4; ++kt) {
    v8h run[2];
#pragma unroll
    for (int rr = 0; rr < 2; ++rr) {
      int k0 = kt * 32 + gg * 16 + rr * 8;  // boundary at 64 is run-aligned
      v8h vv = z8h();
      if (ev) {
        if (k0 < 64)
          vv = load8f(&h[(size_t)ti * 64 + k0]);
        else
          vv = sub8f(&h[(size_t)sv * 64 + (k0 - 64)],
                     &h[(size_t)ti * 64 + (k0 - 64)]);
      }
      run[rr] = vv;
    }
    v16h a = cat16(run[0], run[1]);
#pragma unroll
    for (int nt = 0; nt < 8; ++nt) {
      v16h b = *(const v16h*)&w1T[(nt * 16 + nf) * 128 + kt * 32 + gg * 16];
      acc[nt] = wmma16(a, b, acc[nt]);
    }
  }
#pragma unroll
  for (int nt = 0; nt < 8; ++nt) {
    float bb = b1[nt * 16 + nf];
#pragma unroll
    for (int r = 0; r < 8; ++r) {
      int rr = r + 8 * gg;
      t1s[wv * 2048 + rr * 128 + nt * 16 + nf] =
          (_Float16)frelu(acc[nt][r] + bb);
    }
  }
  __syncthreads();

  // ---- GEMM2: out = relu(t1 @ W2^T + b2), 16x64 ----
  v8f acc2[4];
#pragma unroll
  for (int t = 0; t < 4; ++t) acc2[t] = vz8();

#pragma unroll
  for (int kt = 0; kt < 4; ++kt) {
    v8h lo = *(const v8h*)&t1s[wv * 2048 + m * 128 + kt * 32 + gg * 16];
    v8h hi = *(const v8h*)&t1s[wv * 2048 + m * 128 + kt * 32 + gg * 16 + 8];
    v16h a = cat16(lo, hi);
#pragma unroll
    for (int nt = 0; nt < 4; ++nt) {
      v16h b = *(const v16h*)&w2T[(nt * 16 + nf) * 128 + kt * 32 + gg * 16];
      acc2[nt] = wmma16(a, b, acc2[nt]);
    }
  }
#pragma unroll
  for (int nt = 0; nt < 4; ++nt) {
    float bb = b2[nt * 16 + nf];
#pragma unroll
    for (int r = 0; r < 8; ++r) {
      int ee = tileBase + r + 8 * gg;
      if (ee < NE) {
        int di = ee / 17;
        float val = frelu(acc2[nt][r] + bb);
        atomicMax((unsigned int*)&g[(size_t)di * 64 + nt * 16 + nf],
                  __float_as_uint(val));
      }
    }
  }
}

// ---------------- host orchestration ----------------

extern "C" void kernel_launch(void* const* d_in, const int* in_sizes, int n_in,
                              void* d_out, int out_size, void* d_ws, size_t ws_size,
                              hipStream_t stream) {
  const float* x     = (const float*)d_in[0];
  const int*   eidx  = (const int*)d_in[1];
  const float* eattr = (const float*)d_in[2];
  const float* loc   = (const float*)d_in[3];
  // d_in[4] = k (constant 16, compiled in)

  const float* sg_wm[3] = {(const float*)d_in[5],  (const float*)d_in[9],  (const float*)d_in[13]};
  const float* sg_bm[3] = {(const float*)d_in[6],  (const float*)d_in[10], (const float*)d_in[14]};
  const float* sg_wu[3] = {(const float*)d_in[7],  (const float*)d_in[11], (const float*)d_in[15]};
  const float* sg_bu[3] = {(const float*)d_in[8],  (const float*)d_in[12], (const float*)d_in[16]};
  const float* dg_w1[2] = {(const float*)d_in[17], (const float*)d_in[21]};
  const float* dg_b1[2] = {(const float*)d_in[18], (const float*)d_in[22]};
  const float* dg_w2[2] = {(const float*)d_in[19], (const float*)d_in[23]};
  const float* dg_b2[2] = {(const float*)d_in[20], (const float*)d_in[24]};
  const float* f1_w = (const float*)d_in[25];
  const float* f1_b = (const float*)d_in[26];
  const float* f2_w = (const float*)d_in[27];
  const float* f2_b = (const float*)d_in[28];

  const int N = NN, E = EE;
  const size_t ND = (size_t)N * 64;
  float* ws   = (float*)d_ws;
  float* h1   = ws + 0 * ND;
  float* g1   = ws + 1 * ND;
  float* g2   = ws + 2 * ND;
  float* h2   = ws + 3 * ND;
  float* h3   = ws + 4 * ND;
  float* fu   = ws + 5 * ND;
  float* agg  = ws + 6 * ND;
  int*   knn  = (int*)(ws + 7 * ND);

  const int* srcp = eidx;
  const int* dstp = eidx + E;

  const int zgrid = (int)((ND + 255) / 256);
  const int egrid = (E + 127) / 128;
  const int ngrid = (N + 127) / 128;
  const int dgrid = (NEDYN + 63) / 64;

  // kNN graph (independent of conv chain)
  knn_kernel<<<N, 256, 0, stream>>>(loc, knn, N);

  // static conv 1..3
  const float* hin = x;
  float* houts[3] = {h1, h2, h3};
  for (int l = 0; l < 3; ++l) {
    zero_kernel<<<zgrid, 256, 0, stream>>>(agg, (int)ND);
    static_msg_kernel<<<egrid, 256, 0, stream>>>(hin, eattr, srcp, dstp,
                                                 sg_wm[l], sg_bm[l], agg, E);
    Segs su; su.p[0] = agg; su.p[1] = su.p[2] = su.p[3] = su.p[4] = su.p[5] = agg;
    gemm_relu_kernel<<<ngrid, 256, 0, stream>>>(su, sg_wu[l], sg_bu[l],
                                                houts[l], N, 64, 1);
    hin = houts[l];
  }

  // dynamic conv 1..2
  zero_kernel<<<zgrid, 256, 0, stream>>>(g1, (int)ND);
  dyn_msg_kernel<<<dgrid, 128, 0, stream>>>(h1, knn, dg_w1[0], dg_b1[0],
                                            dg_w2[0], dg_b2[0], g1, N);
  zero_kernel<<<zgrid, 256, 0, stream>>>(g2, (int)ND);
  dyn_msg_kernel<<<dgrid, 128, 0, stream>>>(g1, knn, dg_w1[1], dg_b1[1],
                                            dg_w2[1], dg_b2[1], g2, N);

  // fusion: fuse1 = relu(cat[h1,g1,g2,h2,h3] @ f1_w^T + f1_b)
  Segs c1; c1.p[0] = h1; c1.p[1] = g1; c1.p[2] = g2; c1.p[3] = h2; c1.p[4] = h3; c1.p[5] = h3;
  gemm_relu_kernel<<<ngrid, 256, 0, stream>>>(c1, f1_w, f1_b, fu, N, 320, 1);

  // out = cat[fuse1,h1,g1,g2,h2,h3] @ f2_w^T + f2_b
  Segs c2; c2.p[0] = fu; c2.p[1] = h1; c2.p[2] = g1; c2.p[3] = g2; c2.p[4] = h2; c2.p[5] = h3;
  gemm_relu_kernel<<<ngrid, 256, 0, stream>>>(c2, f2_w, f2_b, (float*)d_out,
                                              N, 384, 0);
}
